// Attention_46909632807566
// MI455X (gfx1250) — matrix operand order repaired and checked by reading, not compiled, not run on hardware
//
#include <hip/hip_runtime.h>

// ---------------- problem constants ----------------
#define S_LEN   2048
#define B_SZ    2
#define H_DIM   2048
#define NHEAD   16
#define HD      128
#define M_ROWS  (S_LEN * B_SZ)          // 4096
#define KDIM    H_DIM                   // 2048
#define KT_STEPS (KDIM / 32)            // 64
#define KV_STEPS (S_LEN / 32)           // 64
// softmax scale folded with log2(e) (exp2 -> v_exp_f32)
#define SOFT_SCALE (0.08838834764831845f * 1.4426950408889634f)

typedef __bf16 bf16;
typedef __bf16 v16bf __attribute__((ext_vector_type(16)));
typedef float  v8f   __attribute__((ext_vector_type(8)));

union Frag  { v16bf v; uint4 u[2]; };
union Pack4 { bf16 e[4]; uint2 u; };
union Pack16{ uint4 u[2]; bf16 e[16]; };
union Pack2 { bf16 e[2]; unsigned u; };
union Pack8 { bf16 e[8]; uint4 u; };

static __device__ __forceinline__ v8f wmma_bf16(v16bf a, v16bf b, v8f c) {
    return __builtin_amdgcn_wmma_f32_16x16x32_bf16(false, a, false, b, (short)0, c,
                                                   false, false);
}

// ---- lane <-> lane^16 exchange (VALU permlane, avoids ds_bpermute) ----
static __device__ __forceinline__ unsigned xor16u(unsigned x) {
#if __has_builtin(__builtin_amdgcn_permlanex16)
    int i = (int)x;
    i = __builtin_amdgcn_permlanex16(i, i, 0x76543210, 0xfedcba98, false, false);
    return (unsigned)i;
#else
    return (unsigned)__shfl_xor((int)x, 16, 32);
#endif
}
static __device__ __forceinline__ float xor16f(float x) {
    unsigned u = __builtin_bit_cast(unsigned, x);
    u = xor16u(u);
    return __builtin_bit_cast(float, u);
}
static __device__ __forceinline__ unsigned pack_bf16(float a, float b) {
    Pack2 p; p.e[0] = (bf16)a; p.e[1] = (bf16)b; return p.u;
}

// ---- Tensor Data Mover (guarded; fallback = manual loader) ----
#if defined(__has_builtin)
#if __has_builtin(__builtin_amdgcn_tensor_load_to_lds) && __has_builtin(__builtin_amdgcn_s_wait_tensorcnt)
#define HAVE_TDM 1
#endif
#endif
#ifndef HAVE_TDM
#define HAVE_TDM 0
#endif
#if __has_include(<hip/amd_detail/amd_gfx1250_TDM.h>)
#define TDM_6ARG 1
#else
#define TDM_6ARG 0
#endif

#if HAVE_TDM
typedef unsigned int u32x4 __attribute__((ext_vector_type(4)));
typedef int          i32x4 __attribute__((ext_vector_type(4)));
typedef int          i32x8 __attribute__((ext_vector_type(8)));

// 2D tile load, data_size = 8B units, LDS row padding via pad fields.
static __device__ __forceinline__ void tdm_load_2d(
    const void* gptr, const void* ldsptr,
    unsigned tileW, unsigned tileH,            // in 8B units / rows
    unsigned tensorW, unsigned tensorH,        // in 8B units / rows
    unsigned strideW,                          // row stride in 8B units
    unsigned padIntervalCode, unsigned padAmountCode)
{
    const unsigned long long ga = (unsigned long long)gptr;
    u32x4 g0;
    g0.x = 1u;                                     // count=1 (valid user D#)
    g0.y = (unsigned)(unsigned long long)ldsptr;   // LDS byte offset = addr[31:0]
    g0.z = (unsigned)ga;                           // global_addr[31:0]
    g0.w = (unsigned)(ga >> 32) | 0x80000000u;     // global_addr[56:32] | type=2
    i32x8 g1;
    g1[0] = (int)((3u << 16) | (1u << 20) |        // data_size=8B, pad_enable
                  (padIntervalCode << 22) | (padAmountCode << 25));
    g1[1] = (int)((tensorW & 0xFFFFu) << 16);                       // tensor_dim0 lo
    g1[2] = (int)((tensorW >> 16) | ((tensorH & 0xFFFFu) << 16));   // td0 hi | td1 lo
    g1[3] = (int)((tensorH >> 16) | (tileW << 16));                 // td1 hi | tile_dim0
    g1[4] = (int)(tileH & 0xFFFFu);                                 // tile_dim1 (tile_dim2=0)
    g1[5] = (int)strideW;                                           // tensor_dim0_stride lo
    g1[6] = 0;
    g1[7] = 0;
    const i32x4 gz = {0, 0, 0, 0};
#if TDM_6ARG
    const i32x8 gz8 = {0, 0, 0, 0, 0, 0, 0, 0};
    __builtin_amdgcn_tensor_load_to_lds(g0, g1, gz, gz, gz8, 0);
#else
    __builtin_amdgcn_tensor_load_to_lds(g0, g1, gz, gz, 0);
#endif
}
#endif  // HAVE_TDM

// ---------------- fp32 -> bf16 cast ----------------
__global__ __launch_bounds__(256) void cast_f32_bf16(const float* __restrict__ src,
                                                     bf16* __restrict__ dst, int n4) {
    int i = blockIdx.x * 256 + threadIdx.x;
    if (i >= n4) return;
    float4 v = ((const float4*)src)[i];
    Pack4 o;
    o.e[0] = (bf16)v.x; o.e[1] = (bf16)v.y; o.e[2] = (bf16)v.z; o.e[3] = (bf16)v.w;
    ((uint2*)dst)[i] = o.u;
}

// ---------------- 128x128 block GEMM, LDS double-buffered ----------------
// C[m,n] = sum_k A[m,k] * W[n,k]    (A row-major MxK, W row-major NxK)
// MODE 0: three weights (blockIdx.z), bf16 output scattered to [b*NH+h][s][d]
// MODE 1: single weight, f32 output row-major [m][n] (== [s,b,h])
template <int MODE>
__global__ __launch_bounds__(256) void gemm128(const bf16* __restrict__ A,
                                               const bf16* __restrict__ W0,
                                               const bf16* __restrict__ W1,
                                               const bf16* __restrict__ W2,
                                               bf16* __restrict__ O0,
                                               bf16* __restrict__ O1,
                                               bf16* __restrict__ O2,
                                               float* __restrict__ OutF) {
    const int tid  = threadIdx.x;
    const int lane = tid & 31;
    const int wave = tid >> 5;
    const int lo   = lane & 15;
    const int hi   = lane >> 4;
    const int wm   = wave >> 2;   // 0..1  (64 rows)
    const int wn   = wave & 3;    // 0..3  (32 cols)
    const int nBase = blockIdx.x * 128;
    const int mBase = blockIdx.y * 128;

    const bf16* Bw = W0;
    bf16* Ob = O0;
    if (MODE == 0) {
        Bw = (blockIdx.z == 0) ? W0 : (blockIdx.z == 1 ? W1 : W2);
        Ob = (blockIdx.z == 0) ? O0 : (blockIdx.z == 1 ? O1 : O2);
    }

    __shared__ __align__(16) bf16 As[2][128][40];   // 32 K + 8 pad (16B pad / 64B row)
    __shared__ __align__(16) bf16 Bs[2][128][40];

    v8f acc[4][2];
    const v8f vzero = {};
#pragma unroll
    for (int i = 0; i < 4; ++i)
#pragma unroll
        for (int j = 0; j < 2; ++j) acc[i][j] = vzero;

#if HAVE_TDM
    if (wave == 0) {
        tdm_load_2d(A  + (size_t)mBase * KDIM, &As[0][0][0], 8, 128, 512, M_ROWS, 512, 3, 3);
        tdm_load_2d(Bw + (size_t)nBase * KDIM, &Bs[0][0][0], 8, 128, 512, H_DIM,  512, 3, 3);
        __builtin_amdgcn_s_wait_tensorcnt(0);
    }
    __syncthreads();
#else
    const int ldRow = tid >> 1;          // 0..127
    const int ldCol = (tid & 1) * 16;    // 0 or 16
    const bf16* Aptr = A  + (size_t)(mBase + ldRow) * KDIM + ldCol;
    const bf16* Bptr = Bw + (size_t)(nBase + ldRow) * KDIM + ldCol;
    {
        uint4 a0 = *(const uint4*)(Aptr);
        uint4 a1 = *(const uint4*)(Aptr + 8);
        uint4 b0 = *(const uint4*)(Bptr);
        uint4 b1 = *(const uint4*)(Bptr + 8);
        *(uint4*)&As[0][ldRow][ldCol]     = a0;
        *(uint4*)&As[0][ldRow][ldCol + 8] = a1;
        *(uint4*)&Bs[0][ldRow][ldCol]     = b0;
        *(uint4*)&Bs[0][ldRow][ldCol + 8] = b1;
    }
    __syncthreads();
#endif

    int buf = 0;
    for (int kt = 0; kt < KT_STEPS; ++kt) {
        const bool pre = (kt + 1 < KT_STEPS);
#if HAVE_TDM
        if (pre && wave == 0) {
            const int kb = (kt + 1) * 32;
            const int nb = buf ^ 1;
            tdm_load_2d(A  + (size_t)mBase * KDIM + kb, &As[nb][0][0], 8, 128, 512, M_ROWS, 512, 3, 3);
            tdm_load_2d(Bw + (size_t)nBase * KDIM + kb, &Bs[nb][0][0], 8, 128, 512, H_DIM,  512, 3, 3);
        }
#else
        uint4 a0, a1, b0, b1;
        if (pre) {
            const int kb = (kt + 1) * 32;
            a0 = *(const uint4*)(Aptr + kb);
            a1 = *(const uint4*)(Aptr + kb + 8);
            b0 = *(const uint4*)(Bptr + kb);
            b1 = *(const uint4*)(Bptr + kb + 8);
        }
#endif
        // fragments from LDS
        v16bf afr[4], bfr[2];
#pragma unroll
        for (int mt = 0; mt < 4; ++mt) {
            Frag f;
            const int ar = wm * 64 + mt * 16 + lo;
            const int ko = hi * 8;
            f.u[0] = *(const uint4*)&As[buf][ar][ko];
            f.u[1] = *(const uint4*)&As[buf][ar][ko + 16];
            afr[mt] = f.v;
        }
#pragma unroll
        for (int nt = 0; nt < 2; ++nt) {
            Frag f;
            const int br = wn * 32 + nt * 16 + lo;
            const int kb2 = hi * 8;
            f.u[0] = *(const uint4*)&Bs[buf][br][kb2];
            f.u[1] = *(const uint4*)&Bs[buf][br][kb2 + 16];
            bfr[nt] = f.v;
        }
#pragma unroll
        for (int mt = 0; mt < 4; ++mt)
#pragma unroll
            for (int nt = 0; nt < 2; ++nt)
                acc[mt][nt] = wmma_bf16(afr[mt], bfr[nt], acc[mt][nt]);

#if HAVE_TDM
        if (pre && wave == 0) __builtin_amdgcn_s_wait_tensorcnt(0);
#else
        if (pre) {
            const int nb = buf ^ 1;
            *(uint4*)&As[nb][ldRow][ldCol]     = a0;
            *(uint4*)&As[nb][ldRow][ldCol + 8] = a1;
            *(uint4*)&Bs[nb][ldRow][ldCol]     = b0;
            *(uint4*)&Bs[nb][ldRow][ldCol + 8] = b1;
        }
#endif
        __syncthreads();
        buf ^= 1;
    }

    // epilogue: C frag -> lane holds N=lo, rows M = g + 8*hi
#pragma unroll
    for (int mt = 0; mt < 4; ++mt)
#pragma unroll
        for (int nt = 0; nt < 2; ++nt)
#pragma unroll
            for (int g = 0; g < 8; ++g) {
                const int m = mBase + wm * 64 + mt * 16 + g + hi * 8;
                const int n = nBase + wn * 32 + nt * 16 + lo;
                const float val = acc[mt][nt][g];
                if (MODE == 0) {
                    const int s = m >> 1, b = m & 1;       // m = s*B + b
                    const int h = n >> 7, d = n & 127;     // n = h*HD + d
                    Ob[((size_t)(b * NHEAD + h) * S_LEN + s) * HD + d] = (bf16)val;
                } else {
                    OutF[(size_t)m * H_DIM + n] = val;
                }
            }
}

// ---------------- flash attention (S^T formulation) ----------------
// Per WG: 128 q rows of one head; per wave: 16 q rows.
// St = K(32 x HD) @ Q^T(HD x 16)  -> q is the N axis (= lane) => per-lane softmax.
// O^T = V^T(HD x 32) @ P(32 x 16) -> packed 16B epilogue stores.
__global__ __launch_bounds__(256) void flash_attn(const bf16* __restrict__ Qh,
                                                  const bf16* __restrict__ Kh,
                                                  const bf16* __restrict__ Vh,
                                                  bf16* __restrict__ Ctx) {
    const int tid  = threadIdx.x;
    const int lane = tid & 31;
    const int wave = tid >> 5;
    const int lo   = lane & 15;
    const int hi   = lane >> 4;
    const int bh    = blockIdx.y;        // b*NH + h
    const int qBase = blockIdx.x * 128;
    const size_t headOff = (size_t)bh * S_LEN * HD;
    const bf16* Qp = Qh + headOff;
    const bf16* Kp = Kh + headOff;
    const bf16* Vp = Vh + headOff;

    __shared__ __align__(16) bf16 Kt[32][136];   // [kv][d], 16B row pad
    __shared__ __align__(16) bf16 Vt[128][40];   // [d][kv] transposed

    // resident Q^T B-operand fragments: lane = q column (lo), d at 8*hi..+7 and 16+8*hi..+7
    v16bf qf[4];
    {
        const int qrow = qBase + wave * 16 + lo;
        const bf16* qrp = Qp + (size_t)qrow * HD;
#pragma unroll
        for (int c = 0; c < 4; ++c) {
            Frag f;
            f.u[0] = *(const uint4*)(qrp + c * 32 + hi * 8);
            f.u[1] = *(const uint4*)(qrp + c * 32 + hi * 8 + 16);
            qf[c] = f.v;
        }
    }

    v8f acc[8];                    // O^T tiles: d rows, q cols(=lane)
    const v8f vzero = {};
#pragma unroll
    for (int dt = 0; dt < 8; ++dt) acc[dt] = vzero;
    float m = -3.0e38f, l = 0.0f;  // per-lane (q) softmax state

    const int krow = tid >> 3;             // 0..31 (kv)
    const int kcol = (tid & 7) * 16;       // 0..112 (d)

    for (int kvt = 0; kvt < KV_STEPS; ++kvt) {
        const int kv0 = kvt * 32;
        __syncthreads();   // previous step's LDS reads done
#if HAVE_TDM
        if (wave == 0)
            tdm_load_2d(Kp + (size_t)kv0 * HD, &Kt[0][0], 32, 32, 32, S_LEN, 32, 5, 3);
#else
        {
            const bf16* krp = Kp + (size_t)(kv0 + krow) * HD + kcol;
            *(uint4*)&Kt[krow][kcol]     = *(const uint4*)krp;
            *(uint4*)&Kt[krow][kcol + 8] = *(const uint4*)(krp + 8);
        }
#endif
        {   // V tile transpose (manual: TDM cannot transpose)
            const bf16* vrp = Vp + (size_t)(kv0 + krow) * HD + kcol;
            Pack16 vv;
            vv.u[0] = *(const uint4*)vrp;
            vv.u[1] = *(const uint4*)(vrp + 8);
#pragma unroll
            for (int j = 0; j < 16; ++j) Vt[kcol + j][krow] = vv.e[j];
        }
#if HAVE_TDM
        if (wave == 0) __builtin_amdgcn_s_wait_tensorcnt(0);
#endif
        __syncthreads();

        // St tiles (16kv x 16q): A = K rows from Kt (A-interleave), B = resident Q^T
        v8f st[2]; st[0] = vzero; st[1] = vzero;
#pragma unroll
        for (int t = 0; t < 2; ++t) {
#pragma unroll
            for (int c = 0; c < 4; ++c) {
                Frag fk;
                const int kr = t * 16 + lo;
                const int dof = c * 32 + hi * 8;
                fk.u[0] = *(const uint4*)&Kt[kr][dof];
                fk.u[1] = *(const uint4*)&Kt[kr][dof + 16];
                st[t] = wmma_bf16(fk.v, qf[c], st[t]);
            }
        }

        // per-lane online softmax over kv (lane pair L, L^16 share q=lo)
        float tmax = -3.0e38f;
#pragma unroll
        for (int t = 0; t < 2; ++t)
#pragma unroll
            for (int g = 0; g < 8; ++g) tmax = fmaxf(tmax, st[t][g]);
        tmax = fmaxf(tmax, xor16f(tmax));
        const float mnew  = fmaxf(m, tmax);
        const float alpha = exp2f((m - mnew) * SOFT_SCALE);
        m = mnew;
        float pf[2][8];
        float r = 0.0f;
#pragma unroll
        for (int t = 0; t < 2; ++t)
#pragma unroll
            for (int g = 0; g < 8; ++g) {
                const float p = exp2f((st[t][g] - mnew) * SOFT_SCALE);
                pf[t][g] = p;
                r += p;
            }
        r += xor16f(r);
        l = l * alpha + r;
#pragma unroll
        for (int dt = 0; dt < 8; ++dt)
#pragma unroll
            for (int g = 0; g < 8; ++g) acc[dt][g] *= alpha;

        // P fragment (B operand, K=kv): own half + permlanex16 partner half
        unsigned pk0[4], pk1[4], po0[4], po1[4];
#pragma unroll
        for (int j = 0; j < 4; ++j) {
            pk0[j] = pack_bf16(pf[0][2 * j], pf[0][2 * j + 1]);
            pk1[j] = pack_bf16(pf[1][2 * j], pf[1][2 * j + 1]);
        }
#pragma unroll
        for (int j = 0; j < 4; ++j) { po0[j] = xor16u(pk0[j]); po1[j] = xor16u(pk1[j]); }
        Frag fb;
        fb.u[0].x = hi ? po1[0] : pk0[0];
        fb.u[0].y = hi ? po1[1] : pk0[1];
        fb.u[0].z = hi ? po1[2] : pk0[2];
        fb.u[0].w = hi ? po1[3] : pk0[3];
        fb.u[1].x = hi ? pk1[0] : po0[0];
        fb.u[1].y = hi ? pk1[1] : po0[1];
        fb.u[1].z = hi ? pk1[2] : po0[2];
        fb.u[1].w = hi ? pk1[3] : po0[3];

        // O^T += V^T @ P : A = Vt rows (contiguous k = 16*hi + s), one wmma per d-tile
#pragma unroll
        for (int dt = 0; dt < 8; ++dt) {
            Frag fv;
            const int d = dt * 16 + lo;
            fv.u[0] = *(const uint4*)&Vt[d][hi * 16];
            fv.u[1] = *(const uint4*)&Vt[d][hi * 16 + 8];
            acc[dt] = wmma_bf16(fv.v, fb.v, acc[dt]);
        }
    }

    // epilogue: lane q=lo, d = dt*16 + 8*hi + g -> 8 contiguous bf16 per store
    const float inv = 1.0f / l;
    const int b = bh >> 4, h = bh & 15;
    const int s = qBase + wave * 16 + lo;
    bf16* crow = Ctx + (size_t)(s * B_SZ + b) * H_DIM + h * HD;
#pragma unroll
    for (int dt = 0; dt < 8; ++dt) {
        Pack8 ov;
#pragma unroll
        for (int g = 0; g < 8; ++g) ov.e[g] = (bf16)(acc[dt][g] * inv);
        *(uint4*)(crow + dt * 16 + 8 * hi) = ov.u;
    }
}

// ---------------- host launcher ----------------
extern "C" void kernel_launch(void* const* d_in, const int* in_sizes, int n_in,
                              void* d_out, int out_size, void* d_ws, size_t ws_size,
                              hipStream_t stream) {
    (void)in_sizes; (void)n_in; (void)out_size; (void)ws_size;
    const float* x  = (const float*)d_in[0];
    const float* wq = (const float*)d_in[1];
    const float* wk = (const float*)d_in[2];
    const float* wv = (const float*)d_in[3];
    const float* wo = (const float*)d_in[4];
    float* out = (float*)d_out;

    char* ws = (char*)d_ws;
    size_t off = 0;
    const size_t xBytes = (size_t)M_ROWS * KDIM * sizeof(bf16);   // 16 MB
    const size_t wBytes = (size_t)H_DIM * H_DIM * sizeof(bf16);   // 8 MB
    bf16* xb  = (bf16*)(ws + off); off += xBytes;
    bf16* wqb = (bf16*)(ws + off); off += wBytes;
    bf16* wkb = (bf16*)(ws + off); off += wBytes;
    bf16* wvb = (bf16*)(ws + off); off += wBytes;
    bf16* wob = (bf16*)(ws + off); off += wBytes;
    bf16* Qh  = (bf16*)(ws + off); off += xBytes;
    bf16* Kh  = (bf16*)(ws + off); off += xBytes;
    bf16* Vh  = (bf16*)(ws + off); off += xBytes;
    bf16* Ctx = (bf16*)(ws + off); off += xBytes;

    {
        const int nx4 = M_ROWS * KDIM / 4;
        const int nw4 = H_DIM * H_DIM / 4;
        cast_f32_bf16<<<(nx4 + 255) / 256, 256, 0, stream>>>(x,  xb,  nx4);
        cast_f32_bf16<<<(nw4 + 255) / 256, 256, 0, stream>>>(wq, wqb, nw4);
        cast_f32_bf16<<<(nw4 + 255) / 256, 256, 0, stream>>>(wk, wkb, nw4);
        cast_f32_bf16<<<(nw4 + 255) / 256, 256, 0, stream>>>(wv, wvb, nw4);
        cast_f32_bf16<<<(nw4 + 255) / 256, 256, 0, stream>>>(wo, wob, nw4);
    }
    {
        dim3 grid(H_DIM / 128, M_ROWS / 128, 3);
        gemm128<0><<<grid, 256, 0, stream>>>(xb, wqb, wkb, wvb, Qh, Kh, Vh, nullptr);
    }
    {
        dim3 grid(S_LEN / 128, B_SZ * NHEAD);
        flash_attn<<<grid, 256, 0, stream>>>(Qh, Kh, Vh, Ctx);
    }
    {
        dim3 grid(H_DIM / 128, M_ROWS / 128, 1);
        gemm128<1><<<grid, 256, 0, stream>>>(Ctx, wob, nullptr, nullptr,
                                             nullptr, nullptr, nullptr, out);
    }
}
